// MagnetohydrodynamicsPhysics_9620726743339
// MI455X (gfx1250) — compile-verified, matrix operand order checked by reading, not run on hardware
//
#include <hip/hip_runtime.h>

typedef __attribute__((ext_vector_type(16))) _Float16 v16h;
typedef __attribute__((ext_vector_type(8)))  _Float16 v8h;
typedef __attribute__((ext_vector_type(2)))  _Float16 v2h;
typedef __attribute__((ext_vector_type(2)))  __fp16   v2hp;  // cvt_pkrtz result type
typedef __attribute__((ext_vector_type(8)))  float    v8f;

// Same-wave LDS ordering fence: CDNA5 split counters; prevents compiler
// reordering (memory clobber) and waits for outstanding DS ops.
#define LDS_FENCE() asm volatile("s_wait_dscnt 0" ::: "memory")

#define WMMA_F16(a, b, c) \
  __builtin_amdgcn_wmma_f32_16x16x32_f16(false, (a), false, (b), (short)0, (c), false, false)

// Weight-fragment pool offsets (in halfs). Each 32x16 (KxN) B-fragment is
// 512 halfs stored lane-major: frag[lane*16 + j].
#define OFF_W1 0        // 1 kstep * 8 ntiles * 512
#define OFF_W2 4096     // 4 * 8 * 512
#define OFF_W3 20480    // 4 * 4 * 512
#define OFF_W4 28672    // 2 * 1 * 512
#define WF_TOTAL 29696

// Fast GELU: gelu(x) ~= x * sigmoid(1.5957691*(x + 0.044715 x^3))
//           = x * E/(E+1) = x - x*rcp(E+1),  E = exp2(x*(K1 + K2*x^2))
// K1 = 1.5957691216*log2(e) = 2.3022083, K2 = K1*0.044715 = 0.1029437
// 5 VALU + 2 TRANS (v_exp_f32 / v_rcp_f32, co-execute with XDL WMMA).
__device__ __forceinline__ float gelu_f(float x) {
  float x2 = x * x;
  float p  = __builtin_fmaf(0.1029437f, x2, 2.3022083f);
  float E  = __builtin_amdgcn_exp2f(x * p);
  float r  = __builtin_amdgcn_rcpf(E + 1.0f);
  return __builtin_fmaf(-x, r, x);   // x - x*r
}

__device__ __forceinline__ v8f splat8(float v) {
  v8f r;
#pragma unroll
  for (int j = 0; j < 8; ++j) r[j] = v;
  return r;
}

// Pack W[K][N] (f32 row-major, fan_in x fan_out) into f16 WMMA B-fragments.
// 16-bit operand striping (ISA 7.12.2): lane<16 -> col=lane, holds K = hi*8+j
// (j<8) and 16+hi*8+(j-8) (j>=8) within each K-step of 32, hi = lane>>4.
// If brow != nullptr, the bias vector is placed at K == Kd (augmented-GEMM
// bias row; the matching activation column is set to 1.0).
__device__ __forceinline__ void pack_frags(const float* __restrict__ W,
                                           const float* __restrict__ brow,
                                           int Kd, int Nd, int KS, int NT,
                                           _Float16* dst, int tid, int nth) {
  int total = KS * NT * 512;
  for (int idx = tid; idx < total; idx += nth) {
    int f  = idx >> 9;
    int e  = idx & 511;
    int ks = f / NT, nt = f % NT;
    int lane = e >> 4, j = e & 15;
    int hi = lane >> 4, col = lane & 15;
    int kk = (j < 8) ? (hi * 8 + j) : (16 + hi * 8 + (j - 8));
    int K  = ks * 32 + kk;
    int n  = nt * 16 + col;
    float v = 0.0f;
    if (n < Nd) {
      if (K < Kd)               v = W[K * Nd + n];
      else if (brow && K == Kd) v = brow[n];
    }
    dst[idx] = (_Float16)v;
  }
}

// Load a 16x32 f16 A-fragment from the wave's activation panel (row-major,
// 128 halfs per row). Lane hi-half selection matches the ISA A layout, so the
// two 8-half chunks per lane are contiguous 16B LDS reads (ds_load_b128).
__device__ __forceinline__ v16h load_afrag(const _Float16* act, int rb, int ks, int lane) {
  int hi = (lane >> 4) & 1;
  int r  = rb + (lane & 15);
  const _Float16* p = act + r * 128 + ks * 32 + hi * 8;
  v8h lo = *(const v8h*)(p);
  v8h h2 = *(const v8h*)(p + 16);
  return __builtin_shufflevector(lo, h2, 0, 1, 2, 3, 4, 5, 6, 7,
                                 8, 9, 10, 11, 12, 13, 14, 15);
}

// GELU + packed f16 writeback of a 16x16 f32 D tile (bias already in C or in
// the augmented weight row — no adds here).
// D layout: element j of v8f at lane -> row = rb + j + 8*(lane>>4), col = lane&15.
// Converts result pairs with v_cvt_pk_rtz_f16_f32.
__device__ __forceinline__ void store_tile_gelu(v8f acc, _Float16* act,
                                                int rb, int colBase, int lane) {
  int hi  = (lane >> 4) & 1;
  int col = colBase + (lane & 15);
  _Float16* base = act + (rb + 8 * hi) * 128 + col;
#pragma unroll
  for (int j = 0; j < 8; j += 2) {
    float g0 = gelu_f(acc[j]);
    float g1 = gelu_f(acc[j + 1]);
    v2hp pkraw = __builtin_amdgcn_cvt_pkrtz(g0, g1);
    v2h pk = __builtin_bit_cast(v2h, pkraw);
    base[(j)     * 128] = pk[0];
    base[(j + 1) * 128] = pk[1];
  }
}

__global__ __launch_bounds__(128) void mhd_mlp_wmma_kernel(
    const float* __restrict__ flow, const float* __restrict__ phys,
    const float* __restrict__ W1, const float* __restrict__ b1,
    const float* __restrict__ W2, const float* __restrict__ b2,
    const float* __restrict__ W3, const float* __restrict__ b3,
    const float* __restrict__ W4, const float* __restrict__ b4,
    float* __restrict__ out, int nBlockIters) {
  // ~99 KB static LDS (gfx1250 WGP has 320 KB; up to 320 KB per workgroup).
  __shared__ __align__(16) _Float16 s_wf[WF_TOTAL];       // 58 KB packed weights
  __shared__ float                  s_bias[336];          // b1|b2|b3|b4(pad16)
  __shared__ __align__(16) _Float16 s_act[4 * 32 * 128];  // 32 KB: per-wave act panel
  __shared__ float                  s_ost[4 * 32 * 16];   // 8 KB: per-wave out staging

  const int tid = threadIdx.x, nth = blockDim.x;
  // W1 carries b1 as augmented row K=6 (feature[6] = 1.0 below).
  pack_frags(W1, b1,        6, 128, 1, 8, s_wf + OFF_W1, tid, nth);
  pack_frags(W2, nullptr, 128, 128, 4, 8, s_wf + OFF_W2, tid, nth);
  pack_frags(W3, nullptr, 128,  64, 4, 4, s_wf + OFF_W3, tid, nth);
  pack_frags(W4, nullptr,  64,   3, 2, 1, s_wf + OFF_W4, tid, nth);
  for (int i = tid; i < 128; i += nth) { s_bias[i] = b1[i]; s_bias[128 + i] = b2[i]; }
  for (int i = tid; i < 64; i += nth)  s_bias[256 + i] = b3[i];
  for (int i = tid; i < 16; i += nth)  s_bias[320 + i] = (i < 3) ? b4[i] : 0.0f;
  __syncthreads();  // only barrier; hot loop is wave-private in LDS

  const int wid  = tid >> 5, lane = tid & 31;
  const int hi   = (lane >> 4) & 1;
  const int col  = lane & 15;
  _Float16* act  = s_act + wid * (32 * 128);
  float*    ost  = s_ost + wid * (32 * 16);
  const int C    = 1 << 18;  // 64^3

  // Persistent bias splat vectors (live across the whole voxel loop): fed to
  // the first WMMA of each tile as the C operand. Since they stay live, the
  // allocator gives WMMA a distinct VDST -> zero per-iteration bias cost.
  v8f cb2[8], cb3[4];
#pragma unroll
  for (int nt = 0; nt < 8; ++nt) cb2[nt] = splat8(s_bias[128 + nt * 16 + col]);
#pragma unroll
  for (int nt = 0; nt < 4; ++nt) cb3[nt] = splat8(s_bias[256 + nt * 16 + col]);
  const float b4s = 0.1f * s_bias[320 + col];
  LDS_FENCE();

  for (int bb = blockIdx.x; bb < nBlockIters; bb += gridDim.x) {
    // ---- pointwise phase: curl(B), Lorentz, f16 features; 1 lane = 1 voxel ----
    int vox = bb * 128 + wid * 32 + lane;
    int bat = vox >> 18;
    int s   = vox & (C - 1);
    int h = s >> 12, w = (s >> 6) & 63, d = s & 63;
    int hp = (h + 1) & 63, hm = (h - 1) & 63;
    int wp = (w + 1) & 63, wm = (w - 1) & 63;
    int dp = (d + 1) & 63, dm = (d - 1) & 63;
    const float* Pb = phys + (((size_t)bat * 3) << 18);
    const float* Vb = flow + (((size_t)bat * 3) << 18);
#define IDX(c, hh, ww, dd) ((c) * C + ((hh) << 12) + ((ww) << 6) + (dd))
    float Bx = Pb[IDX(0, h, w, d)], By = Pb[IDX(1, h, w, d)], Bz = Pb[IDX(2, h, w, d)];
    float Jx = 0.5f * ((Pb[IDX(2, h, wp, d)] - Pb[IDX(2, h, wm, d)]) -
                       (Pb[IDX(1, h, w, dp)] - Pb[IDX(1, h, w, dm)]));
    float Jy = 0.5f * ((Pb[IDX(0, h, w, dp)] - Pb[IDX(0, h, w, dm)]) -
                       (Pb[IDX(2, hp, w, d)] - Pb[IDX(2, hm, w, d)]));
    float Jz = 0.5f * ((Pb[IDX(1, hp, w, d)] - Pb[IDX(1, hm, w, d)]) -
                       (Pb[IDX(0, h, wp, d)] - Pb[IDX(0, h, wm, d)]));
    float Lx = (Jy * Bz - Jz * By) * 2500.0f;  // HARTMANN^2
    float Ly = (Jz * Bx - Jx * Bz) * 2500.0f;
    float Lz = (Jx * By - Jy * Bx) * 2500.0f;
    float vx = Vb[IDX(0, h, w, d)], vy = Vb[IDX(1, h, w, d)], vz = Vb[IDX(2, h, w, d)];
#undef IDX

    union { v8h v; _Float16 e[8]; } f0;
    f0.e[0] = (_Float16)vx; f0.e[1] = (_Float16)vy; f0.e[2] = (_Float16)vz;
    f0.e[3] = (_Float16)Bx; f0.e[4] = (_Float16)By; f0.e[5] = (_Float16)Bz;
    f0.e[6] = (_Float16)1.0f;   // bias column for augmented W1 row K=6
    f0.e[7] = (_Float16)0.0f;
    v8h zz = {};
    *(v8h*)(act + lane * 128 + 0)  = f0.v;  // features cols 0..6, K-pad 7..31 = 0
    *(v8h*)(act + lane * 128 + 8)  = zz;
    *(v8h*)(act + lane * 128 + 16) = zz;
    *(v8h*)(act + lane * 128 + 24) = zz;
    LDS_FENCE();

    // ---- WMMA MLP: wave owns rows [0,32) of its private panel ----
#pragma unroll
    for (int rt = 0; rt < 2; ++rt) {
      const int rb = rt * 16;
      {  // layer 1: [16x32(K=6+bias pad)] x [32x128]; C = inline 0 (bias in W row)
        v16h a = load_afrag(act, rb, 0, lane);
        v8f accs[8];
#pragma unroll
        for (int nt = 0; nt < 8; ++nt) {
          v16h bf = *(const v16h*)(s_wf + OFF_W1 + nt * 512 + lane * 16);
          v8f cz = {};
          accs[nt] = WMMA_F16(a, bf, cz);
        }
        LDS_FENCE();
#pragma unroll
        for (int nt = 0; nt < 8; ++nt)
          store_tile_gelu(accs[nt], act, rb, nt * 16, lane);
        LDS_FENCE();
      }
      {  // layer 2: [16x128] x [128x128] — 32 WMMAs; bias enters via C operand
        v8f accs[8];
#pragma unroll
        for (int ks = 0; ks < 4; ++ks) {
          v16h a = load_afrag(act, rb, ks, lane);
#pragma unroll
          for (int nt = 0; nt < 8; ++nt) {
            v16h bf = *(const v16h*)(s_wf + OFF_W2 + (ks * 8 + nt) * 512 + lane * 16);
            accs[nt] = WMMA_F16(a, bf, (ks == 0) ? cb2[nt] : accs[nt]);
          }
        }
        LDS_FENCE();
#pragma unroll
        for (int nt = 0; nt < 8; ++nt)
          store_tile_gelu(accs[nt], act, rb, nt * 16, lane);
        LDS_FENCE();
      }
      {  // layer 3: [16x128] x [128x64] — 16 WMMAs; bias enters via C operand
        v8f accs[4];
#pragma unroll
        for (int ks = 0; ks < 4; ++ks) {
          v16h a = load_afrag(act, rb, ks, lane);
#pragma unroll
          for (int nt = 0; nt < 4; ++nt) {
            v16h bf = *(const v16h*)(s_wf + OFF_W3 + (ks * 4 + nt) * 512 + lane * 16);
            accs[nt] = WMMA_F16(a, bf, (ks == 0) ? cb3[nt] : accs[nt]);
          }
        }
        LDS_FENCE();
#pragma unroll
        for (int nt = 0; nt < 4; ++nt)
          store_tile_gelu(accs[nt], act, rb, nt * 16, lane);
        LDS_FENCE();
      }
      {  // layer 4: [16x64] x [64x16(N=3 pad)] — 2 WMMAs; out = 0.1*acc + 0.1*b4
        v8f acc;
#pragma unroll
        for (int ks = 0; ks < 2; ++ks) {
          v16h a  = load_afrag(act, rb, ks, lane);
          v16h bf = *(const v16h*)(s_wf + OFF_W4 + ks * 512 + lane * 16);
          if (ks == 0) { v8f cz = {}; acc = WMMA_F16(a, bf, cz); }
          else         { acc = WMMA_F16(a, bf, acc); }
        }
        float* obase = ost + (rb + 8 * hi) * 16 + col;
#pragma unroll
        for (int j = 0; j < 8; ++j)
          obase[j * 16] = __builtin_fmaf(0.1f, acc[j], b4s);
      }
    }
    LDS_FENCE();

    // ---- combine: out = lorentz + 0.1*enhanced (lane reads back its voxel) ----
    float e0 = ost[lane * 16 + 0];
    float e1 = ost[lane * 16 + 1];
    float e2 = ost[lane * 16 + 2];
    size_t ob = ((size_t)bat * 3) << 18;
    out[ob + 0 * (size_t)C + s] = Lx + e0;
    out[ob + 1 * (size_t)C + s] = Ly + e1;
    out[ob + 2 * (size_t)C + s] = Lz + e2;
    LDS_FENCE();
  }
}

extern "C" void kernel_launch(void* const* d_in, const int* in_sizes, int n_in,
                              void* d_out, int out_size, void* d_ws, size_t ws_size,
                              hipStream_t stream) {
  (void)n_in; (void)out_size; (void)d_ws; (void)ws_size;
  const float* flow = (const float*)d_in[0];
  const float* phys = (const float*)d_in[1];
  const float* W1 = (const float*)d_in[2];
  const float* b1 = (const float*)d_in[3];
  const float* W2 = (const float*)d_in[4];
  const float* b2 = (const float*)d_in[5];
  const float* W3 = (const float*)d_in[6];
  const float* b3 = (const float*)d_in[7];
  const float* W4 = (const float*)d_in[8];
  const float* b4 = (const float*)d_in[9];
  float* out = (float*)d_out;

  int totalVox    = in_sizes[0] / 3;            // 2*64^3 = 524288
  int nBlockIters = (totalVox + 127) / 128;     // 128 voxels per block-iteration
  int grid        = nBlockIters < 1024 ? nBlockIters : 1024;
  hipLaunchKernelGGL(mhd_mlp_wmma_kernel, dim3(grid), dim3(128), 0, stream,
                     flow, phys, W1, b1, W2, b2, W3, b3, W4, b4, out, nBlockIters);
}
